// OntoGNN_72507637891700
// MI455X (gfx1250) — matrix operand order repaired and checked by reading, not compiled, not run on hardware
//
#include <hip/hip_runtime.h>
#include <hip/hip_bf16.h>
#include <math.h>

// ---------------- problem constants (match reference) ----------------
#define HID     64
#define HEADS   4
#define N_VAR   500000
#define N_EMP   3000
#define N_SHIFT 30000
#define E_SD    960000
#define E_VE    500000
#define E_VS    500000
#define F_EMP   32
#define F_SHIFT 24
#define F_VAR   19

typedef __attribute__((ext_vector_type(16))) __bf16 v16bf;
typedef __attribute__((ext_vector_type(8)))  float  v8f;
typedef __attribute__((ext_vector_type(4)))  unsigned int u32x4;
typedef __attribute__((ext_vector_type(4)))  float  f32x4;

union Frag {            // 32 bytes: one WMMA 16-bit operand per lane
    v16bf v;
    struct { u32x4 lo, hi; } u;
};

__device__ __forceinline__ float atomicMaxF(float* addr, float val) {
    unsigned int* ua = (unsigned int*)addr;
    unsigned int old = *ua;
    while (true) {
        float f = __uint_as_float(old);
        if (f >= val) break;
        unsigned int assumed = old;
        old = atomicCAS(ua, assumed, __float_as_uint(val));
        if (old == assumed) break;
    }
    return __uint_as_float(old);
}

// ---------------- packing kernels ----------------
// fp32 [M,K] row-major -> bf16 [Mp,Kp], zero padded. Thread = (row, 8-wide k group).
__global__ void pack_a(const float* __restrict__ X, __bf16* __restrict__ A,
                       int M, int K, int Mp, int Kp) {
    int groups = Kp >> 3;
    int t = blockIdx.x * blockDim.x + threadIdx.x;
    if (t >= Mp * groups) return;
    int row = t / groups, g = t - row * groups;
    int kb = g * 8;
    union { __bf16 h[8]; u32x4 u; } tmp;
#pragma unroll
    for (int e = 0; e < 8; ++e) {
        int k = kb + e;
        float f = (row < M && k < K) ? X[(size_t)row * K + k] : 0.f;
        tmp.h[e] = (__bf16)f;
    }
    *(u32x4*)(A + (size_t)row * Kp + kb) = tmp.u;
}

// fp32 W[K,N] -> bf16 fragment-ordered: Bp[((kt*nCT+ct)*32+lane)*16 + e] = W[kt*32+part*8+(e<8?e:e+8)][ct*16+(lane&15)]
__global__ void pack_b(const float* __restrict__ W, __bf16* __restrict__ Bp, int K, int N) {
    int nKT = (K + 31) >> 5, nCT = N >> 4;
    int t = blockIdx.x * blockDim.x + threadIdx.x;
    if (t >= nKT * nCT * 32) return;
    int lane = t & 31;
    int r = t >> 5;
    int ct = r % nCT, kt = r / nCT;
    int part = lane >> 4, col = ct * 16 + (lane & 15);
    union { __bf16 h[16]; u32x4 u[2]; } tmp;
#pragma unroll
    for (int e = 0; e < 16; ++e) {
        int k = kt * 32 + part * 8 + (e < 8 ? e : e + 8);
        tmp.h[e] = (k < K) ? (__bf16)W[(size_t)k * N + col] : (__bf16)0.f;
    }
    u32x4* out = (u32x4*)(Bp + (size_t)t * 16);
    out[0] = tmp.u[0];
    out[1] = tmp.u[1];
}

// ---------------- WMMA GEMM on packed operands ----------------
// Y[M,N](f32) = act(A_bf16[Mp,Kp] @ Bp(+bias)); optional bf16 mirror Ybf[M,N].
__global__ void gemm_wmma(const __bf16* __restrict__ A, const __bf16* __restrict__ Bp,
                          const float* __restrict__ bias, float* __restrict__ Y,
                          __bf16* __restrict__ Ybf, int M, int Kp, int N, int act) {
    int wave = (int)((blockIdx.x * blockDim.x + threadIdx.x) >> 5);
    int lane = threadIdx.x & 31;
    int nCT = N >> 4;
    int rowTile = wave / nCT;
    int ct = wave - rowTile * nCT;
    int rowBase = rowTile * 16;
    if (rowBase >= M) return;                 // wave-uniform
    int part = lane >> 4;
    int row = rowBase + (lane & 15);          // padded rows always readable
    int nKT = Kp >> 5;

    const __bf16* ap = A + (size_t)row * Kp + part * 8;
    const __bf16* bp = Bp + ((size_t)ct * 32 + lane) * 16;
    size_t bstep = (size_t)nCT * 512;

    v8f c = {0.f, 0.f, 0.f, 0.f, 0.f, 0.f, 0.f, 0.f};
    for (int kt = 0; kt < nKT; ++kt) {
        Frag af, bf;
        af.u.lo = *(const u32x4*)(ap);
        af.u.hi = *(const u32x4*)(ap + 16);
        bf.u.lo = *(const u32x4*)(bp);
        bf.u.hi = *(const u32x4*)(bp + 8);
        c = __builtin_amdgcn_wmma_f32_16x16x32_bf16(false, af.v, false, bf.v,
                                                    (short)0, c, false, false);
        ap += 32;
        bp += bstep;
    }

    int col = ct * 16 + (lane & 15);
    float bv = bias ? bias[col] : 0.f;
#pragma unroll
    for (int r = 0; r < 8; ++r) {
        int orow = rowBase + r + 8 * part;
        if (orow < M) {
            float v = c[r] + bv;
            if (act) v = fmaxf(v, 0.f);
            Y[(size_t)orow * N + col] = v;
            if (Ybf) Ybf[(size_t)orow * N + col] = (__bf16)v;
        }
    }
}

// ---------------- misc elementwise / edge kernels ----------------
__global__ void fill_kernel(float* p, size_t n, float v) {
    size_t i = (size_t)blockIdx.x * blockDim.x + threadIdx.x;
    size_t stride = (size_t)gridDim.x * blockDim.x;
    for (; i < n; i += stride) p[i] = v;
}

__global__ void gat_scores(const float* __restrict__ xgat,
                           const float* __restrict__ att_src, const float* __restrict__ att_dst,
                           float* __restrict__ a_s, float* __restrict__ a_d, int n) {
    int i = blockIdx.x * blockDim.x + threadIdx.x;
    if (i >= n * HEADS) return;
    int node = i >> 2, h = i & 3;
    const f32x4* xp = (const f32x4*)(xgat + (size_t)node * 256 + h * 64);
    const f32x4* as = (const f32x4*)(att_src + h * 64);
    const f32x4* ad = (const f32x4*)(att_dst + h * 64);
    float s = 0.f, d = 0.f;
#pragma unroll
    for (int j = 0; j < 16; ++j) {
        f32x4 x = xp[j], u = as[j], w = ad[j];
        s += x.x * u.x + x.y * u.y + x.z * u.z + x.w * u.w;
        d += x.x * w.x + x.y * w.y + x.z * w.z + x.w * w.w;
    }
    a_s[i] = s; a_d[i] = d;
}

__global__ void gat_edge_max(const int* __restrict__ src, const int* __restrict__ dst,
                             const float* __restrict__ a_s, const float* __restrict__ a_d,
                             float* __restrict__ m, int E) {
    int e = blockIdx.x * blockDim.x + threadIdx.x;
    if (e >= E) return;
    int s = src[e], d = dst[e];
#pragma unroll
    for (int h = 0; h < HEADS; ++h) {
        float v = a_s[s * 4 + h] + a_d[d * 4 + h];
        v = (v > 0.f) ? v : 0.2f * v;      // leaky_relu(0.2)
        atomicMaxF(&m[d * 4 + h], v);
    }
}

__global__ void gat_edge_exp(const int* __restrict__ src, const int* __restrict__ dst,
                             const float* __restrict__ a_s, const float* __restrict__ a_d,
                             const float* __restrict__ m, float* __restrict__ exbuf,
                             float* __restrict__ z, int E) {
    int e = blockIdx.x * blockDim.x + threadIdx.x;
    if (e >= E) return;
    int s = src[e], d = dst[e];
#pragma unroll
    for (int h = 0; h < HEADS; ++h) {
        float v = a_s[s * 4 + h] + a_d[d * 4 + h];
        v = (v > 0.f) ? v : 0.2f * v;
        float ex = expf(v - m[d * 4 + h]);
        exbuf[e * 4 + h] = ex;
        atomicAdd(&z[d * 4 + h], ex);
    }
}

__global__ void gat_edge_scatter(const int* __restrict__ src, const int* __restrict__ dst,
                                 const float* __restrict__ xgat, const float* __restrict__ exbuf,
                                 const float* __restrict__ z, float* __restrict__ gacc, int E) {
    int t = blockIdx.x * blockDim.x + threadIdx.x;   // E*64 threads
    int e = t >> 6, dim = t & 63;
    if (e >= E) return;
    int s = src[e], d = dst[e];
#pragma unroll
    for (int h = 0; h < HEADS; ++h) {
        float alpha = exbuf[e * 4 + h] / fmaxf(z[d * 4 + h], 1e-16f);
        atomicAdd(&gacc[(size_t)d * 256 + h * 64 + dim],
                  xgat[(size_t)s * 256 + h * 64 + dim] * alpha);
    }
}

__global__ void gat_finalize(const float* __restrict__ gacc, const float* __restrict__ gat_b,
                             float* __restrict__ h_shift, __bf16* __restrict__ hsb, int n) {
    int i = blockIdx.x * blockDim.x + threadIdx.x;   // n*64 threads
    if (i >= n * HID) return;
    int node = i >> 6, d = i & 63;
    float s = 0.f;
#pragma unroll
    for (int h = 0; h < HEADS; ++h) s += gacc[(size_t)node * 256 + h * 64 + d];
    s = s * 0.25f + gat_b[d];                        // head mean + bias
    float o = fmaxf(s, 0.f) + h_shift[i];            // relu + residual
    h_shift[i] = o;
    hsb[i] = (__bf16)o;                              // refresh bf16 mirror for stage-2 GEMMs
}

__global__ void inj_score(const int* __restrict__ dstv, const int* __restrict__ srce,
                          const float* __restrict__ q, const float* __restrict__ k,
                          float* __restrict__ s_edge, float* __restrict__ m, int E) {
    int e = blockIdx.x * blockDim.x + threadIdx.x;
    if (e >= E) return;
    int d = dstv[e], s = srce[e];
    const f32x4* qp = (const f32x4*)(q + (size_t)d * HID);
    const f32x4* kp = (const f32x4*)(k + (size_t)s * HID);
    float acc = 0.f;
#pragma unroll
    for (int j = 0; j < 16; ++j) {
        f32x4 a = qp[j], b = kp[j];
        acc += a.x * b.x + a.y * b.y + a.z * b.z + a.w * b.w;
    }
    acc *= 0.125f;                 // 1/sqrt(64)
    s_edge[e] = acc;
    atomicMaxF(&m[d], acc);
}

__global__ void inj_exp(const int* __restrict__ dstv, float* __restrict__ s_edge,
                        const float* __restrict__ m, float* __restrict__ z, int E) {
    int e = blockIdx.x * blockDim.x + threadIdx.x;
    if (e >= E) return;
    int d = dstv[e];
    float ex = expf(s_edge[e] - m[d]);
    s_edge[e] = ex;
    atomicAdd(&z[d], ex);
}

__global__ void inj_scatter(const int* __restrict__ dstv, const int* __restrict__ srce,
                            const float* __restrict__ s_edge, const float* __restrict__ z,
                            const float* __restrict__ vals, float* __restrict__ msg, int E) {
    int t = blockIdx.x * blockDim.x + threadIdx.x;   // E*64 threads
    int e = t >> 6, dim = t & 63;
    if (e >= E) return;
    int d = dstv[e], s = srce[e];
    float alpha = s_edge[e] / fmaxf(z[d], 1e-9f);
    atomicAdd(&msg[(size_t)d * HID + dim], vals[(size_t)s * HID + dim] * alpha);
}

// ---------------- fused concat-GEMM + bias + relu + row L2-norm ----------------
// out[m] = || relu([h_var|msg_e|msg_s][m,:192] @ fuseW + fuseb) ||_2 ; fuseW pre-packed bf16.
__global__ void fuse_norm_kernel(const float* __restrict__ hvar, const float* __restrict__ msge,
                                 const float* __restrict__ msgs, const __bf16* __restrict__ Bp,
                                 const float* __restrict__ fuseb, float* __restrict__ out, int M) {
    int wave = (int)((blockIdx.x * blockDim.x + threadIdx.x) >> 5);
    int lane = threadIdx.x & 31;
    int rowBase = wave * 16;
    if (rowBase >= M) return;
    int part = lane >> 4;
    int row = rowBase + (lane & 15);     // M % 16 == 0 -> always valid

    float ss[8] = {0.f, 0.f, 0.f, 0.f, 0.f, 0.f, 0.f, 0.f};

#pragma unroll
    for (int ct = 0; ct < 4; ++ct) {
        v8f c = {0.f, 0.f, 0.f, 0.f, 0.f, 0.f, 0.f, 0.f};
#pragma unroll
        for (int kt = 0; kt < 6; ++kt) {               // K = 192
            int c1 = kt * 32 + part * 8;               // 8-float chunk, never straddles a segment
            int c2 = c1 + 16;
            const float* p1 = (c1 < 64) ? hvar + (size_t)row * 64 + c1
                            : (c1 < 128) ? msge + (size_t)row * 64 + (c1 - 64)
                                         : msgs + (size_t)row * 64 + (c1 - 128);
            const float* p2 = (c2 < 64) ? hvar + (size_t)row * 64 + c2
                            : (c2 < 128) ? msge + (size_t)row * 64 + (c2 - 64)
                                         : msgs + (size_t)row * 64 + (c2 - 128);
            union { f32x4 f[4]; float s[16]; } tf;
            tf.f[0] = ((const f32x4*)p1)[0];
            tf.f[1] = ((const f32x4*)p1)[1];
            tf.f[2] = ((const f32x4*)p2)[0];
            tf.f[3] = ((const f32x4*)p2)[1];
            v16bf a;
#pragma unroll
            for (int e = 0; e < 16; ++e) a[e] = (__bf16)tf.s[e];
            const __bf16* bp = Bp + (((size_t)kt * 4 + ct) * 32 + lane) * 16;
            Frag bf;
            bf.u.lo = *(const u32x4*)(bp);
            bf.u.hi = *(const u32x4*)(bp + 8);
            c = __builtin_amdgcn_wmma_f32_16x16x32_bf16(false, a, false, bf.v,
                                                        (short)0, c, false, false);
        }
        int col = ct * 16 + (lane & 15);
        float bv = fuseb[col];
#pragma unroll
        for (int r = 0; r < 8; ++r) {
            float v = fmaxf(c[r] + bv, 0.f);
            ss[r] += v * v;
        }
    }
#pragma unroll
    for (int r = 0; r < 8; ++r) {        // reduce over 16-lane half (rows' columns)
        float v = ss[r];
        v += __shfl_xor(v, 1, 32);
        v += __shfl_xor(v, 2, 32);
        v += __shfl_xor(v, 4, 32);
        v += __shfl_xor(v, 8, 32);
        ss[r] = v;
    }
    if ((lane & 15) == 0) {
#pragma unroll
        for (int r = 0; r < 8; ++r) {
            int orow = rowBase + r + 8 * part;
            if (orow < M) out[orow] = sqrtf(ss[r]);
        }
    }
}

// ---------------- host launcher ----------------
static inline long cdivl(long a, long b) { return (a + b - 1) / b; }

extern "C" void kernel_launch(void* const* d_in, const int* in_sizes, int n_in,
                              void* d_out, int out_size, void* d_ws, size_t ws_size,
                              hipStream_t stream) {
    const float* x_emp    = (const float*)d_in[0];
    const float* x_shift  = (const float*)d_in[1];
    const float* x_var    = (const float*)d_in[2];
    // d_in[3] = x_con : dead downstream, skipped
    const float* W_emp    = (const float*)d_in[4];
    const float* b_emp    = (const float*)d_in[5];
    const float* W_shift  = (const float*)d_in[6];
    const float* b_shift  = (const float*)d_in[7];
    const float* W_var    = (const float*)d_in[8];
    const float* b_var    = (const float*)d_in[9];
    const float* gat_W    = (const float*)d_in[12];
    const float* att_src  = (const float*)d_in[13];
    const float* att_dst  = (const float*)d_in[14];
    const float* gat_b    = (const float*)d_in[15];
    const float* injW_emp = (const float*)d_in[16];
    const float* injb_emp = (const float*)d_in[17];
    const float* injW_sh  = (const float*)d_in[18];
    const float* injb_sh  = (const float*)d_in[19];
    const float* qW_emp   = (const float*)d_in[20];
    const float* kW_emp   = (const float*)d_in[21];
    const float* qW_sh    = (const float*)d_in[22];
    const float* kW_sh    = (const float*)d_in[23];
    const float* fuseW    = (const float*)d_in[24];
    const float* fuseb    = (const float*)d_in[25];
    const int*   sd_src   = (const int*)d_in[26];
    const int*   sd_dst   = (const int*)d_in[27];
    const int*   ve_var   = (const int*)d_in[28];
    const int*   ve_emp   = (const int*)d_in[29];
    const int*   vs_var   = (const int*)d_in[30];
    const int*   vs_shift = (const int*)d_in[31];
    (void)in_sizes; (void)n_in; (void)out_size; (void)ws_size;

    // byte allocator over d_ws, 256B aligned
    char* base = (char*)d_ws;
    size_t off = 0;
    auto alloc = [&](size_t bytes) -> void* {
        off = (off + 255) & ~(size_t)255;
        void* p = base + off;
        off += bytes;
        return p;
    };
    const int MP_EMP = 3008;   // 3000 rounded to 16
    // fp32 buffers
    float* h_emp   = (float*)alloc((size_t)N_EMP * HID * 4);
    float* h_shift = (float*)alloc((size_t)N_SHIFT * HID * 4);
    float* h_var   = (float*)alloc((size_t)N_VAR * HID * 4);
    float* xgat    = (float*)alloc((size_t)N_SHIFT * 256 * 4);
    float* a_s     = (float*)alloc((size_t)N_SHIFT * HEADS * 4);
    float* a_d     = (float*)alloc((size_t)N_SHIFT * HEADS * 4);
    float* m_gat   = (float*)alloc((size_t)N_SHIFT * HEADS * 4);
    float* z_gat   = (float*)alloc((size_t)N_SHIFT * HEADS * 4);
    float* gacc    = (float*)alloc((size_t)N_SHIFT * 256 * 4);
    float* exbuf   = (float*)alloc((size_t)E_SD * HEADS * 4);
    float* vals    = (float*)alloc((size_t)N_SHIFT * HID * 4);
    float* kbuf    = (float*)alloc((size_t)N_SHIFT * HID * 4);
    float* q       = (float*)alloc((size_t)N_VAR * HID * 4);
    float* s_edge  = (float*)alloc((size_t)E_VE * 4);
    float* m_var   = (float*)alloc((size_t)N_VAR * 4);
    float* z_var   = (float*)alloc((size_t)N_VAR * 4);
    float* msg_e   = (float*)alloc((size_t)N_VAR * HID * 4);
    float* msg_s   = (float*)alloc((size_t)N_VAR * HID * 4);
    // bf16 packed A buffers
    __bf16* xemp_p   = (__bf16*)alloc((size_t)MP_EMP * 32 * 2);
    __bf16* xshift_p = (__bf16*)alloc((size_t)N_SHIFT * 32 * 2);
    __bf16* xvar_p   = (__bf16*)alloc((size_t)N_VAR * 32 * 2);
    __bf16* hemp_b   = (__bf16*)alloc((size_t)MP_EMP * HID * 2);
    __bf16* hshift_b = (__bf16*)alloc((size_t)N_SHIFT * HID * 2);
    __bf16* hvar_b   = (__bf16*)alloc((size_t)N_VAR * HID * 2);

    const int BLK = 256;
    auto packA = [&](const float* X, __bf16* A, int M, int K, int Mp, int Kp) {
        long thr = (long)Mp * (Kp / 8);
        pack_a<<<dim3((unsigned)cdivl(thr, BLK)), dim3(BLK), 0, stream>>>(X, A, M, K, Mp, Kp);
    };
    auto packB = [&](const float* W, int K, int N) -> __bf16* {
        int nKT = (K + 31) / 32, nCT = N / 16;
        __bf16* p = (__bf16*)alloc((size_t)nKT * nCT * 512 * 2);
        long thr = (long)nKT * nCT * 32;
        pack_b<<<dim3((unsigned)cdivl(thr, BLK)), dim3(BLK), 0, stream>>>(W, p, K, N);
        return p;
    };
    auto gemm = [&](const __bf16* A, const __bf16* Bp, const float* bias, float* Y,
                    __bf16* Ybf, int M, int Kp, int N, int act) {
        long waves = cdivl(M, 16) * (N / 16);
        long blocks = cdivl(waves * 32, BLK);
        gemm_wmma<<<dim3((unsigned)blocks), dim3(BLK), 0, stream>>>(A, Bp, bias, Y, Ybf,
                                                                    M, Kp, N, act);
    };
    auto fill = [&](float* p, size_t n, float v) {
        long b = cdivl((long)n, BLK);
        unsigned blocks = (unsigned)(b < 4096 ? b : 4096);
        if (blocks == 0) blocks = 1;
        fill_kernel<<<dim3(blocks), dim3(BLK), 0, stream>>>(p, n, v);
    };
    const float NEG_INF = -3.4e38f;

    // weight packing (tiny one-off kernels)
    __bf16* Wb_emp  = packB(W_emp,   F_EMP,   HID);
    __bf16* Wb_sh   = packB(W_shift, F_SHIFT, HID);
    __bf16* Wb_var  = packB(W_var,   F_VAR,   HID);
    __bf16* Wb_gat  = packB(gat_W,   HID,     256);
    __bf16* Wb_inje = packB(injW_emp, HID, HID);
    __bf16* Wb_ke   = packB(kW_emp,   HID, HID);
    __bf16* Wb_qe   = packB(qW_emp,   HID, HID);
    __bf16* Wb_injs = packB(injW_sh,  HID, HID);
    __bf16* Wb_ks   = packB(kW_sh,    HID, HID);
    __bf16* Wb_qs   = packB(qW_sh,    HID, HID);
    __bf16* Wb_fuse = packB(fuseW,    192, HID);

    // Stage 0: pack inputs, project (relu(x@W+b)); h_con dead -> skipped
    packA(x_emp,   xemp_p,   N_EMP,   F_EMP,   MP_EMP,  32);
    packA(x_shift, xshift_p, N_SHIFT, F_SHIFT, N_SHIFT, 32);
    packA(x_var,   xvar_p,   N_VAR,   F_VAR,   N_VAR,   32);
    gemm(xemp_p,   Wb_emp, b_emp,   h_emp,   hemp_b,   N_EMP,   32, HID, 1);
    gemm(xshift_p, Wb_sh,  b_shift, h_shift, hshift_b, N_SHIFT, 32, HID, 1);
    gemm(xvar_p,   Wb_var, b_var,   h_var,   hvar_b,   N_VAR,   32, HID, 1);

    // Stage 1: GAT on shift / same-day edges
    gemm(hshift_b, Wb_gat, nullptr, xgat, nullptr, N_SHIFT, HID, 256, 0);
    gat_scores<<<dim3((unsigned)cdivl((long)N_SHIFT * HEADS, BLK)), dim3(BLK), 0, stream>>>(
        xgat, att_src, att_dst, a_s, a_d, N_SHIFT);
    fill(m_gat, (size_t)N_SHIFT * HEADS, NEG_INF);
    fill(z_gat, (size_t)N_SHIFT * HEADS, 0.f);
    fill(gacc, (size_t)N_SHIFT * 256, 0.f);
    gat_edge_max<<<dim3((unsigned)cdivl(E_SD, BLK)), dim3(BLK), 0, stream>>>(
        sd_src, sd_dst, a_s, a_d, m_gat, E_SD);
    gat_edge_exp<<<dim3((unsigned)cdivl(E_SD, BLK)), dim3(BLK), 0, stream>>>(
        sd_src, sd_dst, a_s, a_d, m_gat, exbuf, z_gat, E_SD);
    gat_edge_scatter<<<dim3((unsigned)cdivl((long)E_SD * 64, BLK)), dim3(BLK), 0, stream>>>(
        sd_src, sd_dst, xgat, exbuf, z_gat, gacc, E_SD);
    gat_finalize<<<dim3((unsigned)cdivl((long)N_SHIFT * HID, BLK)), dim3(BLK), 0, stream>>>(
        gacc, gat_b, h_shift, hshift_b, N_SHIFT);

    // Stage 2a: inject emp -> var
    gemm(hemp_b, Wb_inje, injb_emp, vals, nullptr, N_EMP, HID, HID, 0);
    gemm(hemp_b, Wb_ke,   nullptr,  kbuf, nullptr, N_EMP, HID, HID, 0);
    gemm(hvar_b, Wb_qe,   nullptr,  q,    nullptr, N_VAR, HID, HID, 0);
    fill(m_var, (size_t)N_VAR, NEG_INF);
    fill(z_var, (size_t)N_VAR, 0.f);
    fill(msg_e, (size_t)N_VAR * HID, 0.f);
    inj_score<<<dim3((unsigned)cdivl(E_VE, BLK)), dim3(BLK), 0, stream>>>(
        ve_var, ve_emp, q, kbuf, s_edge, m_var, E_VE);
    inj_exp<<<dim3((unsigned)cdivl(E_VE, BLK)), dim3(BLK), 0, stream>>>(
        ve_var, s_edge, m_var, z_var, E_VE);
    inj_scatter<<<dim3((unsigned)cdivl((long)E_VE * 64, BLK)), dim3(BLK), 0, stream>>>(
        ve_var, ve_emp, s_edge, z_var, vals, msg_e, E_VE);

    // Stage 2b: inject shift -> var (reuse q, s_edge, m_var, z_var, vals, kbuf)
    gemm(hshift_b, Wb_injs, injb_sh, vals, nullptr, N_SHIFT, HID, HID, 0);
    gemm(hshift_b, Wb_ks,   nullptr, kbuf, nullptr, N_SHIFT, HID, HID, 0);
    gemm(hvar_b,   Wb_qs,   nullptr, q,    nullptr, N_VAR,   HID, HID, 0);
    fill(m_var, (size_t)N_VAR, NEG_INF);
    fill(z_var, (size_t)N_VAR, 0.f);
    fill(msg_s, (size_t)N_VAR * HID, 0.f);
    inj_score<<<dim3((unsigned)cdivl(E_VS, BLK)), dim3(BLK), 0, stream>>>(
        vs_var, vs_shift, q, kbuf, s_edge, m_var, E_VS);
    inj_exp<<<dim3((unsigned)cdivl(E_VS, BLK)), dim3(BLK), 0, stream>>>(
        vs_var, s_edge, m_var, z_var, E_VS);
    inj_scatter<<<dim3((unsigned)cdivl((long)E_VS * 64, BLK)), dim3(BLK), 0, stream>>>(
        vs_var, vs_shift, s_edge, z_var, vals, msg_s, E_VS);

    // Stage 3: fused concat-GEMM + bias + relu + per-row L2 norm -> d_out[500000]
    {
        long waves = cdivl(N_VAR, 16);
        long blocks = cdivl(waves * 32, BLK);
        fuse_norm_kernel<<<dim3((unsigned)blocks), dim3(BLK), 0, stream>>>(
            h_var, msg_e, msg_s, Wb_fuse, fuseb, (float*)d_out, N_VAR);
    }
}
